// MLASimilarity_40286793236905
// MI455X (gfx1250) — matrix operand order repaired and checked by reading, not compiled, not run on hardware
//
#include <hip/hip_runtime.h>
#include <hip/hip_bf16.h>

typedef _Float16 half_t;
typedef __attribute__((ext_vector_type(16))) _Float16 v16h;
typedef __attribute__((ext_vector_type(8)))  float    v8f;

#define NB 512
#define NC 256
#define NS 7
#define S2 49
#define NHEAD 8
#define HDIM 32
#define MROWS (NB * S2)   // 25088 = 196 * 128
#define BM 128
#define BN 64
#define BK 32

// ---------------------------------------------------------------------------
// Implicit-im2col WMMA GEMM: Out[M,N] = epi( A_im2col[M,K] * Wt[K,N] + bias )
// A is NHWC f16 activations (B,49,Cin); Wt is K-major f16 (K = Cin*ksz*ksz).
// LDS B tile is stored transposed (Bs[n][k]) so B fragments are contiguous.
// EPI: 0 = f16 store, 1 = relu+f16, 2 = routed += mask[b,e]*v (f16 RMW),
//      3 = f32 NCHW store (final output).
// ---------------------------------------------------------------------------
template<int EPI>
__global__ __launch_bounds__(256)
void conv_wmma(const half_t* __restrict__ Ain, int Cin, int ksz, int pad,
               const half_t* __restrict__ Wt, const float* __restrict__ bias,
               int N, void* __restrict__ OutP,
               const float* __restrict__ scale, int expertIdx)
{
    __shared__ __align__(16) half_t As[BM][BK + 8];   // [m][k]
    __shared__ __align__(16) half_t Bs[BN][BK + 8];   // [n][k] (transposed)

    const int tid  = threadIdx.x;
    const int lane = tid & 31;
    const int wid  = tid >> 5;
    const int wm0  = (wid & 3) * 32;
    const int wn0  = (wid >> 2) * 32;
    const int nbase = blockIdx.x * BN;
    const int rbase = blockIdx.y * BM;
    const int k2 = ksz * ksz;
    const int K  = Cin * k2;

    v8f acc[2][2];
#pragma unroll
    for (int a = 0; a < 2; a++)
#pragma unroll
        for (int b = 0; b < 2; b++)
            acc[a][b] = (v8f){0.f,0.f,0.f,0.f,0.f,0.f,0.f,0.f};

    const int mrow = lane & 15;
    const int hi8  = (lane >> 4) << 3;   // 0 or 8
    const int hi16 = (lane >> 4) << 4;   // 0 or 16
    const int ncol = lane & 15;

    for (int k0 = 0; k0 < K; k0 += BK) {
        // prefetch next weight tile into L2 (global_prefetch_b8)
        if (k0 + BK < K) {
            const half_t* nxt = Wt + (size_t)(k0 + BK + (tid >> 3)) * N + nbase + ((tid & 7) << 3);
            __builtin_prefetch((const void*)nxt, 0, 1);
        }

        // ---- stage A tile (implicit im2col, f16) ----
        if (ksz == 1) {
            // contiguous channels: 16B vector copies
            for (int i = tid; i < BM * BK / 8; i += 256) {
                int mt = i >> 2, seg = i & 3;
                int r = rbase + mt;
                int b = r / S2, p = r - b * S2;
                float4 v = *(const float4*)&Ain[(size_t)(b * S2 + p) * Cin + k0 + seg * 8];
                *(float4*)&As[mt][seg * 8] = v;
            }
        } else {
            for (int i = tid; i < BM * BK; i += 256) {
                int mt = i >> 5, kt = i & 31;
                int r  = rbase + mt;
                int b  = r / S2;  int p = r - b * S2;
                int y  = p / NS;  int x = p - y * NS;
                int kk = k0 + kt;
                int c  = kk / k2; int rem = kk - c * k2;
                int kh = rem / ksz, kw = rem - kh * ksz;
                int iy = y + kh - pad, ix = x + kw - pad;
                half_t v = (half_t)0.f;
                if (iy >= 0 && iy < NS && ix >= 0 && ix < NS)
                    v = Ain[(size_t)(b * S2 + iy * NS + ix) * Cin + c];
                As[mt][kt] = v;
            }
        }
        // ---- stage B tile transposed: Bs[n][k] ----
        {
            int kt = tid >> 3, seg = tid & 7;             // 32 x 8 = 256 threads
            float4 v = *(const float4*)&Wt[(size_t)(k0 + kt) * N + nbase + seg * 8];
            const half_t* hp = (const half_t*)&v;
#pragma unroll
            for (int j = 0; j < 8; j++)
                Bs[seg * 8 + j][kt] = hp[j];
        }
        __syncthreads();

        // ---- fragments + WMMA ----
        v16h af[2], bf[2];
#pragma unroll
        for (int a = 0; a < 2; a++) {
            int m = wm0 + 16 * a + mrow;
#pragma unroll
            for (int h = 0; h < 16; h++)
                af[a][h] = As[m][(h < 8 ? h : h + 8) + hi8];
        }
#pragma unroll
        for (int bb = 0; bb < 2; bb++) {
            int n = wn0 + 16 * bb + ncol;
#pragma unroll
            for (int h = 0; h < 16; h++)
                bf[bb][h] = Bs[n][(h < 8 ? h : h + 8) + hi8];
        }
#pragma unroll
        for (int a = 0; a < 2; a++)
#pragma unroll
            for (int bb = 0; bb < 2; bb++)
                acc[a][bb] = __builtin_amdgcn_wmma_f32_16x16x32_f16(
                    false, af[a], false, bf[bb], (short)0, acc[a][bb], false, false);
        __syncthreads();
    }

    // ---- epilogue ----
#pragma unroll
    for (int a = 0; a < 2; a++) {
#pragma unroll
        for (int bb = 0; bb < 2; bb++) {
#pragma unroll
            for (int r0 = 0; r0 < 8; r0++) {
                int row = rbase + wm0 + 16 * a + r0 + hi8;
                int col = nbase + wn0 + 16 * bb + ncol;
                float v = acc[a][bb][r0] + bias[col];
                if (EPI == 0) {
                    ((half_t*)OutP)[(size_t)row * N + col] = (half_t)v;
                } else if (EPI == 1) {
                    ((half_t*)OutP)[(size_t)row * N + col] = (half_t)(v > 0.f ? v : 0.f);
                } else if (EPI == 2) {
                    int b = row / S2;
                    float s = scale[b * 4 + expertIdx];
                    half_t* o = (half_t*)OutP;
                    size_t idx = (size_t)row * N + col;
                    o[idx] = (half_t)((float)o[idx] + s * v);
                } else {
                    int b = row / S2, p = row - b * S2;
                    ((float*)OutP)[((size_t)b * NC + col) * S2 + p] = v;
                }
            }
        }
    }
}

// ---------------------------------------------------------------------------
// Attention: one block (128 thr = 4 waves) per (b, head). 49 tokens padded to
// 64. Raw scores via WMMA -> softmax (bias + mask folded in, one thread/row)
// -> out = attn*V via WMMA. V is stored transposed in LDS for contiguous
// fragment loads.
// ---------------------------------------------------------------------------
__global__ __launch_bounds__(128)
void attn_wmma(const half_t* __restrict__ qh, const half_t* __restrict__ kvh,
               const float* __restrict__ posb, half_t* __restrict__ ao)
{
    __shared__ __align__(16) half_t Qs[64][36];     // [pos][d]
    __shared__ __align__(16) half_t Ks[64][36];     // [pos][d]
    __shared__ __align__(16) half_t Vst[32][72];    // [d][pos] (transposed)
    __shared__ float  Af[64][64];
    __shared__ __align__(16) half_t Ah[64][68];

    const int b = blockIdx.x >> 3, h = blockIdx.x & 7;
    const int tid = threadIdx.x, lane = tid & 31, w = tid >> 5;

    for (int i = tid; i < 64 * HDIM; i += 128) {
        int p = i >> 5, d = i & 31;
        half_t qv = (half_t)0.f, kv = (half_t)0.f, vv = (half_t)0.f;
        if (p < S2) {
            size_t base = (size_t)(b * S2 + p);
            qv = qh[base * 256 + h * HDIM + d];
            kv = kvh[base * 512 + h * HDIM + d];
            vv = kvh[base * 512 + 256 + h * HDIM + d];
        }
        Qs[p][d] = qv; Ks[p][d] = kv; Vst[d][p] = vv;
    }
    __syncthreads();

    const int mrow = lane & 15;
    const int hi8  = (lane >> 4) << 3;
    const int hi16 = (lane >> 4) << 4;
    const int ncol = lane & 15;

    // raw scores: wave w owns query rows [16w, 16w+16); scatter unconditionally
    v16h qf;
#pragma unroll
    for (int hh = 0; hh < 16; hh++)
        qf[hh] = Qs[16 * w + mrow][(hh < 8 ? hh : hh + 8) + hi8];
#pragma unroll
    for (int ct = 0; ct < 4; ct++) {
        v16h kf;  // B = K^T : B[d][key] = Ks[key][d] (contiguous in d)
#pragma unroll
        for (int hh = 0; hh < 16; hh++)
            kf[hh] = Ks[16 * ct + ncol][(hh < 8 ? hh : hh + 8) + hi8];
        v8f c = (v8f){0.f,0.f,0.f,0.f,0.f,0.f,0.f,0.f};
        c = __builtin_amdgcn_wmma_f32_16x16x32_f16(false, qf, false, kf,
                                                   (short)0, c, false, false);
#pragma unroll
        for (int r0 = 0; r0 < 8; r0++)
            Af[16 * w + r0 + hi8][16 * ct + ncol] = c[r0];
    }
    __syncthreads();

    // softmax: one thread per row; scale + pos-bias + padding mask folded here
    if (tid < 64) {
        const int row = tid;
        const bool valid = row < S2;
        const float* pb = posb + ((size_t)h * S2 + (valid ? row : 0)) * S2;
        float mx = -1e30f;
        for (int j = 0; j < 64; j++) {
            float L = (valid && j < S2)
                      ? Af[row][j] * 0.17677669529663687f + pb[j]
                      : -1e30f;
            Af[row][j] = L;
            mx = fmaxf(mx, L);
        }
        float sum = 0.f;
        for (int j = 0; j < 64; j++) sum += __expf(Af[row][j] - mx);
        float inv = 1.f / sum;
        for (int j = 0; j < 64; j++)
            Ah[row][j] = (half_t)(__expf(Af[row][j] - mx) * inv);
    }
    __syncthreads();

    // out = attn(64x64) * V(64x32): K = 64 -> 2 steps
    v8f o[2];
    o[0] = (v8f){0.f,0.f,0.f,0.f,0.f,0.f,0.f,0.f};
    o[1] = o[0];
#pragma unroll
    for (int s = 0; s < 2; s++) {
        v16h afr;
#pragma unroll
        for (int hh = 0; hh < 16; hh++)
            afr[hh] = Ah[16 * w + mrow][(hh < 8 ? hh : hh + 8) + hi8 + 32 * s];
#pragma unroll
        for (int nt = 0; nt < 2; nt++) {
            v16h vf;  // B[k=pos][n=d] = Vst[d][pos] (contiguous in pos)
#pragma unroll
            for (int hh = 0; hh < 16; hh++)
                vf[hh] = Vst[16 * nt + ncol][(hh < 8 ? hh : hh + 8) + hi8 + 32 * s];
            o[nt] = __builtin_amdgcn_wmma_f32_16x16x32_f16(false, afr, false, vf,
                                                           (short)0, o[nt], false, false);
        }
    }
#pragma unroll
    for (int nt = 0; nt < 2; nt++)
#pragma unroll
        for (int r0 = 0; r0 < 8; r0++) {
            int qr = 16 * w + r0 + hi8;
            int d  = 16 * nt + ncol;
            if (qr < S2)
                ao[(size_t)(b * S2 + qr) * 256 + h * HDIM + d] = (half_t)o[nt][r0];
        }
}

// ---------------------------------------------------------------------------
// GroupNorm (groups=4) in-place on NHWC f16 buffer; one block per sample.
// ---------------------------------------------------------------------------
__global__ __launch_bounds__(256)
void gn_kernel(half_t* __restrict__ buf, int Cq,
               const float* __restrict__ gamma, const float* __restrict__ beta)
{
    __shared__ float rs[256], rq[256];
    __shared__ float gmu[4], grv[4];
    const int b = blockIdx.x, tid = threadIdx.x;
    const int g = tid >> 6, t = tid & 63;
    const int gc = Cq >> 2;
    const int n  = S2 * gc;
    float s = 0.f, q = 0.f;
    for (int i = t; i < n; i += 64) {
        int p = i / gc, c = g * gc + (i - (i / gc) * gc);
        float v = (float)buf[(size_t)(b * S2 + p) * Cq + c];
        s += v; q += v * v;
    }
    rs[tid] = s; rq[tid] = q;
    __syncthreads();
    for (int off = 32; off > 0; off >>= 1) {
        if (t < off) { rs[tid] += rs[tid + off]; rq[tid] += rq[tid + off]; }
        __syncthreads();
    }
    if (t == 0) {
        float mu  = rs[tid] / (float)n;
        float var = rq[tid] / (float)n - mu * mu;
        gmu[g] = mu; grv[g] = rsqrtf(var + 1e-5f);
    }
    __syncthreads();
    float mu = gmu[g], rv = grv[g];
    for (int i = t; i < n; i += 64) {
        int p = i / gc, c = g * gc + (i - (i / gc) * gc);
        size_t idx = (size_t)(b * S2 + p) * Cq + c;
        float v = (float)buf[idx];
        buf[idx] = (half_t)((v - mu) * rv * gamma[c] + beta[c]);
    }
}

// ---------------------------------------------------------------------------
// Top-2 routing mask: block per sample (input is NCHW f32).
// ---------------------------------------------------------------------------
__global__ __launch_bounds__(256)
void route_kernel(const float* __restrict__ xin, const float* __restrict__ rw,
                  const float* __restrict__ rb, float* __restrict__ mask)
{
    __shared__ float means[256];
    __shared__ float red[256];
    __shared__ float sc[4];
    const int b = blockIdx.x, tid = threadIdx.x;
    const float* px = xin + ((size_t)b * NC + tid) * S2;
    float s = 0.f;
    for (int p = 0; p < S2; p++) s += px[p];
    means[tid] = s / (float)S2;
    __syncthreads();
    const int e = tid >> 6, t = tid & 63;
    float d = 0.f;
    for (int c = t; c < NC; c += 64) d += means[c] * rw[e * NC + c];
    red[tid] = d;
    __syncthreads();
    for (int off = 32; off > 0; off >>= 1) {
        if (t < off) red[tid] += red[tid + off];
        __syncthreads();
    }
    if (t == 0) sc[e] = red[tid] + rb[e];
    __syncthreads();
    if (tid == 0) {
        int i1 = 0;
        for (int j = 1; j < 4; j++) if (sc[j] > sc[i1]) i1 = j;
        int i2 = -1;
        for (int j = 0; j < 4; j++) if (j != i1 && (i2 < 0 || sc[j] > sc[i2])) i2 = j;
        float m  = fmaxf(sc[i1], sc[i2]);
        float e1 = __expf(sc[i1] - m), e2 = __expf(sc[i2] - m);
        float inv = 1.f / (e1 + e2);
        float out[4] = {0.f, 0.f, 0.f, 0.f};
        out[i1] = e1 * inv; out[i2] = e2 * inv;
        for (int j = 0; j < 4; j++) mask[b * 4 + j] = out[j];
    }
}

// ---------------------------------------------------------------------------
// helpers
// ---------------------------------------------------------------------------
__global__ void nchw_to_nhwc_h(const float* __restrict__ src, half_t* __restrict__ dst, int C)
{
    size_t i = (size_t)blockIdx.x * 256 + threadIdx.x;
    size_t total = (size_t)NB * C * S2;
    if (i >= total) return;
    int c = (int)(i % C); size_t r = i / C;
    int p = (int)(r % S2); int b = (int)(r / S2);
    dst[i] = (half_t)src[((size_t)b * C + c) * S2 + p];
}

__global__ void prep_w(const float* __restrict__ src, half_t* __restrict__ dst,
                       int O, int I, int k2)
{
    int i = blockIdx.x * 256 + threadIdx.x;
    int total = O * I * k2;
    if (i >= total) return;
    int o = i / (I * k2); int rem = i - o * (I * k2);
    int ci = rem / k2;    int kk = rem - ci * k2;
    dst[(size_t)(ci * k2 + kk) * O + o] = (half_t)src[i];
}

__global__ void build_posbias(const float* __restrict__ rpb, float* __restrict__ posb)
{
    int i = blockIdx.x * 256 + threadIdx.x;
    if (i >= NHEAD * S2 * S2) return;
    int h = i / (S2 * S2); int rem = i - h * (S2 * S2);
    int qi = rem / S2, ki = rem - (rem / S2) * S2;
    int qy = qi / NS, qx = qi % NS, ky = ki / NS, kx = ki % NS;
    int ry = qy - ky + NS - 1, rx = qx - kx + NS - 1;
    posb[i] = rpb[(h * 13 + ry) * 13 + rx];
}

__global__ void zero_h(half_t* __restrict__ p, size_t n)
{
    size_t i = (size_t)blockIdx.x * 256 + threadIdx.x;
    if (i < n) p[i] = (half_t)0.f;
}

// ---------------------------------------------------------------------------
extern "C" void kernel_launch(void* const* d_in, const int* in_sizes, int n_in,
                              void* d_out, int out_size, void* d_ws, size_t ws_size,
                              hipStream_t stream)
{
    const float* z    = (const float*)d_in[0];
    const float* x    = (const float*)d_in[1];
    const float* rpb  = (const float*)d_in[2];
    const float* tw1  = (const float*)d_in[3];
    const float* tb1  = (const float*)d_in[4];
    const float* tg_g = (const float*)d_in[5];
    const float* tg_b = (const float*)d_in[6];
    const float* tw2  = (const float*)d_in[7];
    const float* tb2  = (const float*)d_in[8];
    const float* xw1  = (const float*)d_in[9];
    const float* xb1  = (const float*)d_in[10];
    const float* xg_g = (const float*)d_in[11];
    const float* xg_b = (const float*)d_in[12];
    const float* xw2  = (const float*)d_in[13];
    const float* xb2  = (const float*)d_in[14];
    const float* rw   = (const float*)d_in[15];
    const float* rb   = (const float*)d_in[16];
    const float* ew1  = (const float*)d_in[17];
    const float* eb1  = (const float*)d_in[18];
    const float* ew2  = (const float*)d_in[19];
    const float* eb2  = (const float*)d_in[20];
    const float* fw1  = (const float*)d_in[21];
    const float* fb1  = (const float*)d_in[22];
    const float* fw2  = (const float*)d_in[23];
    const float* fb2  = (const float*)d_in[24];
    (void)in_sizes; (void)n_in; (void)out_size; (void)ws_size;

    char* ws = (char*)d_ws;
    size_t off = 0;
    auto alloc = [&](size_t bytes) -> void* {
        void* p = ws + off;
        off = (off + bytes + 255) & ~(size_t)255;
        return p;
    };

    half_t* zh   = (half_t*)alloc((size_t)MROWS * 256 * 2);
    half_t* xh   = (half_t*)alloc((size_t)MROWS * 256 * 2);
    half_t* h16  = (half_t*)alloc((size_t)MROWS * 128 * 2);
    half_t* rz   = (half_t*)alloc((size_t)MROWS * 256 * 2);
    half_t* rx   = (half_t*)alloc((size_t)MROWS * 256 * 2);
    half_t* t1h  = (half_t*)alloc((size_t)MROWS * 64 * 2);
    half_t* x1h  = (half_t*)alloc((size_t)MROWS * 128 * 2);
    half_t* qh   = (half_t*)alloc((size_t)MROWS * 256 * 2);
    half_t* kvh  = (half_t*)alloc((size_t)MROWS * 512 * 2);
    half_t* ao   = (half_t*)alloc((size_t)MROWS * 256 * 2);
    half_t* f1   = (half_t*)alloc((size_t)MROWS * 256 * 2);
    float* maskz = (float*)alloc((size_t)NB * 4 * 4);
    float* maskx = (float*)alloc((size_t)NB * 4 * 4);
    float* posb  = (float*)alloc((size_t)NHEAD * S2 * S2 * 4);
    half_t* ew1t = (half_t*)alloc((size_t)4 * 2304 * 128 * 2);
    half_t* ew2t = (half_t*)alloc((size_t)4 * 128 * 256 * 2);
    half_t* tw1t = (half_t*)alloc((size_t)2304 * 64 * 2);
    half_t* tw2t = (half_t*)alloc((size_t)64 * 256 * 2);
    half_t* xw1t = (half_t*)alloc((size_t)2304 * 128 * 2);
    half_t* xw2t = (half_t*)alloc((size_t)128 * 512 * 2);
    half_t* fw1t = (half_t*)alloc((size_t)256 * 256 * 2);
    half_t* fw2t = (half_t*)alloc((size_t)2304 * 256 * 2);

    auto cdiv = [](size_t a, size_t b) { return (unsigned)((a + b - 1) / b); };
    const dim3 blk(256);

    // weight prep (f32 OIHW -> f16 K x N)
    for (int e = 0; e < 4; e++) {
        prep_w<<<cdiv(128 * 256 * 9, 256), blk, 0, stream>>>(
            ew1 + (size_t)e * 128 * 256 * 9, ew1t + (size_t)e * 2304 * 128, 128, 256, 9);
        prep_w<<<cdiv(256 * 128, 256), blk, 0, stream>>>(
            ew2 + (size_t)e * 256 * 128, ew2t + (size_t)e * 128 * 256, 256, 128, 1);
    }
    prep_w<<<cdiv(64 * 256 * 9, 256), blk, 0, stream>>>(tw1, tw1t, 64, 256, 9);
    prep_w<<<cdiv(256 * 64, 256),     blk, 0, stream>>>(tw2, tw2t, 256, 64, 1);
    prep_w<<<cdiv(128 * 256 * 9, 256),blk, 0, stream>>>(xw1, xw1t, 128, 256, 9);
    prep_w<<<cdiv(512 * 128, 256),    blk, 0, stream>>>(xw2, xw2t, 512, 128, 1);
    prep_w<<<cdiv(256 * 256, 256),    blk, 0, stream>>>(fw1, fw1t, 256, 256, 1);
    prep_w<<<cdiv(256 * 256 * 9, 256),blk, 0, stream>>>(fw2, fw2t, 256, 256, 9);
    build_posbias<<<cdiv(NHEAD * S2 * S2, 256), blk, 0, stream>>>(rpb, posb);

    // activations -> NHWC f16; routing masks
    nchw_to_nhwc_h<<<cdiv((size_t)MROWS * 256, 256), blk, 0, stream>>>(z, zh, 256);
    nchw_to_nhwc_h<<<cdiv((size_t)MROWS * 256, 256), blk, 0, stream>>>(x, xh, 256);
    route_kernel<<<NB, blk, 0, stream>>>(z, rw, rb, maskz);
    route_kernel<<<NB, blk, 0, stream>>>(x, rw, rb, maskx);
    zero_h<<<cdiv((size_t)MROWS * 256, 256), blk, 0, stream>>>(rz, (size_t)MROWS * 256);
    zero_h<<<cdiv((size_t)MROWS * 256, 256), blk, 0, stream>>>(rx, (size_t)MROWS * 256);

    // MoE experts (all 4, masked accumulate like the reference)
    for (int e = 0; e < 4; e++) {
        const half_t* w1 = ew1t + (size_t)e * 2304 * 128;
        const half_t* w2 = ew2t + (size_t)e * 128 * 256;
        conv_wmma<1><<<dim3(2, 196), blk, 0, stream>>>(zh, 256, 3, 1, w1, eb1 + e * 128, 128, h16, nullptr, 0);
        conv_wmma<2><<<dim3(4, 196), blk, 0, stream>>>(h16, 128, 1, 0, w2, eb2 + e * 256, 256, rz, maskz, e);
        conv_wmma<1><<<dim3(2, 196), blk, 0, stream>>>(xh, 256, 3, 1, w1, eb1 + e * 128, 128, h16, nullptr, 0);
        conv_wmma<2><<<dim3(4, 196), blk, 0, stream>>>(h16, 128, 1, 0, w2, eb2 + e * 256, 256, rx, maskx, e);
    }

    // q path
    conv_wmma<0><<<dim3(1, 196), blk, 0, stream>>>(rz, 256, 3, 1, tw1t, tb1, 64, t1h, nullptr, 0);
    gn_kernel<<<NB, blk, 0, stream>>>(t1h, 64, tg_g, tg_b);
    conv_wmma<0><<<dim3(4, 196), blk, 0, stream>>>(t1h, 64, 1, 0, tw2t, tb2, 256, qh, nullptr, 0);

    // kv path
    conv_wmma<0><<<dim3(2, 196), blk, 0, stream>>>(rx, 256, 3, 1, xw1t, xb1, 128, x1h, nullptr, 0);
    gn_kernel<<<NB, blk, 0, stream>>>(x1h, 128, xg_g, xg_b);
    conv_wmma<0><<<dim3(8, 196), blk, 0, stream>>>(x1h, 128, 1, 0, xw2t, xb2, 512, kvh, nullptr, 0);

    // attention
    attn_wmma<<<NB * NHEAD, dim3(128), 0, stream>>>(qh, kvh, posb, ao);

    // output head
    conv_wmma<1><<<dim3(4, 196), blk, 0, stream>>>(ao, 256, 1, 0, fw1t, fb1, 256, f1, nullptr, 0);
    conv_wmma<3><<<dim3(4, 196), blk, 0, stream>>>(f1, 256, 3, 1, fw2t, fb2, 256, d_out, nullptr, 0);
}
